// DGP_82300163326729
// MI455X (gfx1250) — matrix operand order repaired and checked by reading, not compiled, not run on hardware
//
#include <hip/hip_runtime.h>
#include <hip/hip_bf16.h>

typedef __attribute__((ext_vector_type(16))) __bf16 v16bf;
typedef __attribute__((ext_vector_type(8)))  float  v8f;
typedef __attribute__((ext_vector_type(4)))  int    v4i;

#define SNr   20480      // S*N = 5*4096
#define Nn    4096
#define Dh    64
#define Mi    128
#define Ssamp 5

__device__ __forceinline__ v8f wmma_bf16(v16bf a, v16bf b, v8f c) {
  return __builtin_amdgcn_wmma_f32_16x16x32_bf16(false, a, false, b, (short)0, c, false, false);
}

// ---------- fragment loaders (CDNA5 16-bit WMMA layouts, ISA 7.12.2) ----------
// A (16x32): lanes 0-15 hold M=lane, K interleave: VGPR0..3 -> K {0..7}+8*half,
// VGPR4..7 -> K {16..23}+8*half where half = lane>>4.
__device__ __forceinline__ v16bf ldA_rm(const __bf16* p, int ld) {
  int lane = threadIdx.x & 31, m = lane & 15, kh = (lane >> 4) & 1;
  v16bf a;
#pragma unroll
  for (int v = 0; v < 8; ++v) {
    int kb = ((v < 4) ? 2 * v : 16 + 2 * (v - 4)) + 8 * kh;
    a[2 * v]     = p[m * ld + kb];
    a[2 * v + 1] = p[m * ld + kb + 1];
  }
  return a;
}
// A fragment where logical element (i,k) lives at p[k*ld + i] (transposed source)
__device__ __forceinline__ v16bf ldA_cm(const __bf16* p, int ld) {
  int lane = threadIdx.x & 31, m = lane & 15, kh = (lane >> 4) & 1;
  v16bf a;
#pragma unroll
  for (int v = 0; v < 8; ++v) {
    int kb = ((v < 4) ? 2 * v : 16 + 2 * (v - 4)) + 8 * kh;
    a[2 * v]     = p[kb * ld + m];
    a[2 * v + 1] = p[(kb + 1) * ld + m];
  }
  return a;
}
// B (32x16): column N=lane&15, same K interleave as A: K {0..7}+8*half, then {16..23}+8*half.
__device__ __forceinline__ v16bf ldB_km(const __bf16* p, int ld) {  // B[k][n] at p[k*ld+n]
  int lane = threadIdx.x & 31, n = lane & 15, kh = (lane >> 4) & 1;
  v16bf b;
#pragma unroll
  for (int v = 0; v < 8; ++v) {
    int kb = ((v < 4) ? 2 * v : 16 + 2 * (v - 4)) + 8 * kh;
    b[2 * v]     = p[kb * ld + n];
    b[2 * v + 1] = p[(kb + 1) * ld + n];
  }
  return b;
}
__device__ __forceinline__ v16bf ldB_nk(const __bf16* p, int ld) {  // B[k][n] at p[n*ld+k]
  int lane = threadIdx.x & 31, n = lane & 15, kh = (lane >> 4) & 1;
  v16bf b;
#pragma unroll
  for (int v = 0; v < 8; ++v) {
    int kb = ((v < 4) ? 2 * v : 16 + 2 * (v - 4)) + 8 * kh;
    b[2 * v]     = p[n * ld + kb];
    b[2 * v + 1] = p[n * ld + kb + 1];
  }
  return b;
}
// C (16x16 f32): VGPR r holds row r + 8*(lane>=16), col = lane&15
__device__ __forceinline__ void stC_bf(__bf16* p, int ld, v8f c) {
  int lane = threadIdx.x & 31, n = lane & 15, mh = (lane >> 4) & 1;
#pragma unroll
  for (int r = 0; r < 8; ++r) p[(r + 8 * mh) * ld + n] = (__bf16)c[r];
}

// ---------- async staging of a 128x16 bf16 tile (row stride ldg) into LDS ----------
// gfx1250 path: GLOBAL_LOAD_ASYNC_TO_LDS_B128 (ASYNCcnt) + s_wait_asynccnt + barrier.
__device__ __forceinline__ void stage_tile_128x16(const __bf16* gsrc, int ldg, __bf16* lds) {
  int t = threadIdx.x;  // blockDim.x == 128; 256 x 16B transfers, 2 per thread
#pragma unroll
  for (int i = 0; i < 2; ++i) {
    int tt = t + i * 128;
    int row = tt >> 1, half = tt & 1;
    const __bf16* g = gsrc + row * ldg + half * 8;
    __bf16* l = lds + row * 16 + half * 8;
#if __has_builtin(__builtin_amdgcn_global_load_async_to_lds_b128)
    __builtin_amdgcn_global_load_async_to_lds_b128(
        (__attribute__((address_space(1))) v4i*)((v4i*)g),
        (__attribute__((address_space(3))) v4i*)((v4i*)l), 0, 0);
#else
    *(v4i*)l = *(const v4i*)g;   // fallback: b128 global load + ds_store_b128
#endif
  }
#if __has_builtin(__builtin_amdgcn_global_load_async_to_lds_b128)
#if __has_builtin(__builtin_amdgcn_s_wait_asynccnt)
  __builtin_amdgcn_s_wait_asynccnt(0);
#else
  asm volatile("s_wait_asynccnt 0x0" ::: "memory");
#endif
#endif
  __syncthreads();
}

// ---------- deterministic gaussian (hash + Box-Muller) ----------
__device__ __forceinline__ unsigned pcg(unsigned x) {
  x = x * 747796405u + 2891336453u;
  unsigned w = ((x >> ((x >> 28) + 4u)) ^ x) * 277803737u;
  return (w >> 22) ^ w;
}
__device__ __forceinline__ float u01(unsigned h) {
  return (float)(h >> 8) * (1.0f / 16777216.0f) + (0.5f / 16777216.0f);
}
__device__ __forceinline__ float gauss(unsigned seed) {
  float u1 = u01(pcg(seed));
  float u2 = u01(pcg(seed ^ 0x9e3779b9u));
  return sqrtf(-2.0f * logf(u1)) * cosf(6.28318530718f * u2);
}

// ---------- small fp32 kernels ----------
__global__ void k_tileX(const float* X, float* F, __bf16* Fb) {
  int i = blockIdx.x * blockDim.x + threadIdx.x;        // SN*Dh exact
  float v = X[i % (Nn * Dh)];
  F[i] = v; Fb[i] = (__bf16)v;
}
__global__ void k_convert(const float* s, __bf16* d, int n) {
  for (int i = blockIdx.x * blockDim.x + threadIdx.x; i < n; i += gridDim.x * blockDim.x)
    d[i] = (__bf16)s[i];
}
__global__ void k_rownorm(const float* X, float* out, int rows) {
  int r = blockIdx.x * blockDim.x + threadIdx.x;
  if (r >= rows) return;
  float s = 0.f;
  for (int d = 0; d < Dh; ++d) { float v = X[r * Dh + d]; s += v * v; }
  out[r] = s;
}
__global__ void k_kmm(const float* Z, float* A, float diagadd) {
  int i = blockIdx.y * 16 + threadIdx.y, j = blockIdx.x * 16 + threadIdx.x;
  float s = 0.f;
  for (int d = 0; d < Dh; ++d) { float t = Z[i * Dh + d] - Z[j * Dh + d]; s += t * t; }
  A[i * Mi + j] = expf(-0.5f * s) + ((i == j) ? diagadd : 0.f);
}
__global__ void k_cholesky(float* A) {  // 1 block, 128 threads, in-place lower
  int i = threadIdx.x;
  for (int j = 0; j < Mi; ++j) {
    if (i == j) {
      float s = A[j * Mi + j];
      for (int k = 0; k < j; ++k) { float l = A[j * Mi + k]; s -= l * l; }
      A[j * Mi + j] = sqrtf(fmaxf(s, 1e-20f));
    }
    __syncthreads();
    float dj = A[j * Mi + j];
    if (i > j) {
      float s = A[i * Mi + j];
      for (int k = 0; k < j; ++k) s -= A[i * Mi + k] * A[j * Mi + k];
      A[i * Mi + j] = s / dj;
    } else if (i < j) {
      A[i * Mi + j] = 0.f;
    }
    __syncthreads();
  }
}
__global__ void k_trinv(const float* L, float* Li) {  // explicit lower-tri inverse, thread=column
  int j = threadIdx.x;
  for (int i = 0; i < Mi; ++i) {
    if (i >= j) {
      float s = (i == j) ? 1.f : 0.f;
      for (int k = j; k < i; ++k) s -= L[i * Mi + k] * Li[k * Mi + j];
      Li[i * Mi + j] = s / L[i * Mi + i];
    } else {
      Li[i * Mi + j] = 0.f;
    }
  }
}
__global__ void k_make_G(const float* qs, __bf16* G, int Dout) {  // G[d,k,m] = tril(qs)[m,k,d]
  int total = Dout * Mi * Mi;
  for (int idx = blockIdx.x * blockDim.x + threadIdx.x; idx < total; idx += gridDim.x * blockDim.x) {
    int d = idx / (Mi * Mi), rem = idx % (Mi * Mi), k = rem / Mi, m = rem % Mi;
    float v = (m >= k) ? qs[(m * Mi + k) * Dout + d] : 0.f;
    G[idx] = (__bf16)v;
  }
}
__global__ void k_fvar0(const __bf16* A, float* fvar0, float c0) {
  int sn = blockIdx.x * blockDim.x + threadIdx.x;  // SN exact
  float s = 0.f;
  for (int m = 0; m < Mi; ++m) { float a = (float)A[m * SNr + sn]; s += a * a; }
  fvar0[sn] = c0 - s;
}
__global__ void k_matvec(const __bf16* A, const float* qmu, float* Fmean) {
  int sn = blockIdx.x * blockDim.x + threadIdx.x;  // SN exact
  float s = 0.f;
  for (int m = 0; m < Mi; ++m) s += (float)A[m * SNr + sn] * qmu[m];
  Fmean[sn] = s;
}

// ---------- WMMA kernels (128 threads = 4 waves per block) ----------
__global__ void k_knm(const __bf16* Fb, const __bf16* Zb, const float* xn2,
                      const float* z2, __bf16* Kt) {
  int wid = (blockIdx.x * blockDim.x + threadIdx.x) >> 5;
  int snt = wid % (SNr / 16), mt = wid / (SNr / 16);
  v8f acc{0.f,0.f,0.f,0.f,0.f,0.f,0.f,0.f};
#pragma unroll
  for (int ks = 0; ks < Dh / 32; ++ks) {
    v16bf a = ldA_rm(Fb + (snt * 16) * Dh + ks * 32, Dh);
    v16bf b = ldB_nk(Zb + (mt * 16) * Dh + ks * 32, Dh);   // B = Z^T chunk
    acc = wmma_bf16(a, b, acc);
  }
  int lane = threadIdx.x & 31, n = lane & 15, mh = (lane >> 4) & 1;
  int m = mt * 16 + n;
  float zz = z2[m];
#pragma unroll
  for (int r = 0; r < 8; ++r) {
    int sn = snt * 16 + r + 8 * mh;
    float v = expf(-0.5f * (xn2[sn] + zz - 2.0f * acc[r]));
    Kt[m * SNr + sn] = (__bf16)v;   // store transposed: Knm^T
  }
}
// A = Linv * Kt ; one block per sn-tile: stage the Kt column tile in LDS once,
// 4 waves cover the 8 row-tiles (2 each).
__global__ void k_Asolve(const __bf16* Linv, const __bf16* Kt, __bf16* Aout) {
  __shared__ __bf16 Bls[Mi * 16];
  int snt = blockIdx.x;
  stage_tile_128x16(Kt + snt * 16, SNr, Bls);
  int w = threadIdx.x >> 5;
  v16bf bfr[Mi / 32];
#pragma unroll
  for (int ms = 0; ms < Mi / 32; ++ms) bfr[ms] = ldB_km(Bls + (ms * 32) * 16, 16);
#pragma unroll
  for (int rr = 0; rr < 2; ++rr) {
    int rt = w + rr * 4;
    v8f acc{0.f,0.f,0.f,0.f,0.f,0.f,0.f,0.f};
#pragma unroll
    for (int ks = 0; ks < Mi / 32; ++ks)
      acc = wmma_bf16(ldA_rm(Linv + (rt * 16) * Mi + ks * 32, Mi), bfr[ks], acc);
    stC_bf(Aout + (rt * 16) * SNr + snt * 16, SNr, acc);
  }
}
// fvar_q: block = (sn-tile, group of 4 output dims); A tile staged once in LDS,
// each wave streams its own G_d (row-major, b128-vectorizable).
__global__ void k_fvarq(const __bf16* G, const __bf16* A, float* fvarq, int Dout) {
  __shared__ __bf16 Bls[Mi * 16];
  int snt = blockIdx.x;
  stage_tile_128x16(A + snt * 16, SNr, Bls);
  int w = threadIdx.x >> 5;
  int d = blockIdx.y * 4 + w;
  if (d >= Dout) return;
  v16bf bfr[Mi / 32];
#pragma unroll
  for (int ms = 0; ms < Mi / 32; ++ms) bfr[ms] = ldB_km(Bls + (ms * 32) * 16, 16);
  const __bf16* Gd = G + (size_t)d * Mi * Mi;
  float tot = 0.f;
#pragma unroll 1
  for (int kt = 0; kt < Mi / 16; ++kt) {
    v8f acc{0.f,0.f,0.f,0.f,0.f,0.f,0.f,0.f};
#pragma unroll
    for (int ms = 0; ms < Mi / 32; ++ms)
      acc = wmma_bf16(ldA_rm(Gd + (kt * 16) * Mi + ms * 32, Mi), bfr[ms], acc);
#pragma unroll
    for (int r = 0; r < 8; ++r) tot += acc[r] * acc[r];
  }
  tot += __shfl_xor(tot, 16, 32);  // combine the two row-halves of each column
  int lane = threadIdx.x & 31;
  if (lane < 16) fvarq[(size_t)(snt * 16 + lane) * Dout + d] = tot;
}
// mean = A^T q_mu + F W ; then sample. One block per sn-tile, A tile staged in LDS,
// the 4 waves are the 4 output-dim tiles.
__global__ void k_mean_sample(const __bf16* A, const __bf16* qmub, const __bf16* Fb_in,
                              const __bf16* Wb, const float* fvar0, const float* fvarq,
                              float* Fout, __bf16* Fout_b, unsigned seed) {
  __shared__ __bf16 Als[Mi * 16];
  int snt = blockIdx.x;
  stage_tile_128x16(A + snt * 16, SNr, Als);
  int dt = threadIdx.x >> 5;
  v8f acc{0.f,0.f,0.f,0.f,0.f,0.f,0.f,0.f};
#pragma unroll
  for (int ks = 0; ks < Mi / 32; ++ks) {                      // A^T * q_mu (A^T from LDS)
    v16bf a = ldA_cm(Als + (ks * 32) * 16, 16);
    v16bf b = ldB_km(qmub + (ks * 32) * Dh + dt * 16, Dh);
    acc = wmma_bf16(a, b, acc);
  }
#pragma unroll
  for (int ks = 0; ks < Dh / 32; ++ks) {                      // + F * W (linear mean fn)
    v16bf a = ldA_rm(Fb_in + (snt * 16) * Dh + ks * 32, Dh);
    v16bf b = ldB_km(Wb + (ks * 32) * Dh + dt * 16, Dh);
    acc = wmma_bf16(a, b, acc);
  }
  int lane = threadIdx.x & 31, n = lane & 15, mh = (lane >> 4) & 1;
  int d = dt * 16 + n;
#pragma unroll
  for (int r = 0; r < 8; ++r) {
    int sn = snt * 16 + r + 8 * mh;
    float var = fvar0[sn] + fvarq[(size_t)sn * Dh + d];
    float eps = gauss(seed ^ (unsigned)(sn * Dh + d));
    float f = acc[r] + sqrtf(fmaxf(var, 1e-12f)) * eps;
    Fout[sn * Dh + d] = f;
    Fout_b[sn * Dh + d] = (__bf16)f;
  }
}

// ---------- deterministic reductions ----------
__global__ void k_ve(const float* Y, const float* Fmean, const float* fv0,
                     const float* fvq, float* pv) {            // 80 blocks x 256 = SN
  int sn = blockIdx.x * blockDim.x + threadIdx.x;
  int nidx = sn % Nn;
  float fm = Fmean[sn], fv = fv0[sn] + fvq[sn];
  float dY = Y[nidx] - fm;
  float v = (-0.91893853320467274f - 0.5f * (dY * dY + fv)) * (1.0f / (float)Ssamp);
  __shared__ float sm[256];
  sm[threadIdx.x] = v; __syncthreads();
  for (int s = 128; s > 0; s >>= 1) {
    if (threadIdx.x < s) sm[threadIdx.x] += sm[threadIdx.x + s];
    __syncthreads();
  }
  if (threadIdx.x == 0) pv[blockIdx.x] = sm[0];
}
__global__ void k_kl(const float* qmu, const float* qs, int Dout, float* pk) {  // 64 blocks x 256
  float local = 0.f;
  int t1 = Mi * Mi * Dout;
  for (int idx = blockIdx.x * blockDim.x + threadIdx.x; idx < t1; idx += gridDim.x * blockDim.x) {
    int t = idx / Dout, k = t % Mi, m = t / Mi;
    float v = qs[idx];
    if (m > k)       local += 0.5f * v * v;
    else if (m == k) local += 0.5f * v * v - logf(fabsf(v) + 1e-30f);
  }
  int t2 = Mi * Dout;
  for (int idx = blockIdx.x * blockDim.x + threadIdx.x; idx < t2; idx += gridDim.x * blockDim.x)
    local += 0.5f * qmu[idx] * qmu[idx];
  __shared__ float sm[256];
  sm[threadIdx.x] = local; __syncthreads();
  for (int s = 128; s > 0; s >>= 1) {
    if (threadIdx.x < s) sm[threadIdx.x] += sm[threadIdx.x + s];
    __syncthreads();
  }
  if (threadIdx.x == 0) pk[blockIdx.x] = sm[0];
}
__global__ void k_finalize(const float* pv, const float* pk, float* out) {
  float L = 0.f;  for (int i = 0; i < 80;  ++i) L += pv[i];
  float KL = 0.f; for (int i = 0; i < 192; ++i) KL += pk[i];
  // + 0.5*M*D per layer constants: 0.5*(128*64 + 128*64 + 128) = 8256
  out[0] = L - KL + 8256.0f;
}

// ---------- host ----------
static inline char* carve(char*& p, size_t bytes) {
  char* r = p; p += (bytes + 255) & ~(size_t)255; return r;
}

extern "C" void kernel_launch(void* const* d_in, const int* in_sizes, int n_in,
                              void* d_out, int out_size, void* d_ws, size_t ws_size,
                              hipStream_t stream) {
  (void)in_sizes; (void)n_in; (void)out_size; (void)ws_size;
  const float* X     = (const float*)d_in[0];
  const float* Y     = (const float*)d_in[1];
  const float* Z_h   = (const float*)d_in[2];
  const float* qmu_h = (const float*)d_in[3];
  const float* qsq_h = (const float*)d_in[4];
  const float* W_h   = (const float*)d_in[5];
  const float* Z_f   = (const float*)d_in[6];
  const float* qmu_f = (const float*)d_in[7];
  const float* qsq_f = (const float*)d_in[8];
  float* out = (float*)d_out;

  char* p = (char*)d_ws;
  float*  pv     = (float*)carve(p, 128 * sizeof(float));
  float*  pk     = (float*)carve(p, 256 * sizeof(float));
  float*  Fcur   = (float*)carve(p, (size_t)SNr * Dh * sizeof(float));
  float*  Fnxt   = (float*)carve(p, (size_t)SNr * Dh * sizeof(float));
  __bf16* Fcurb  = (__bf16*)carve(p, (size_t)SNr * Dh * sizeof(__bf16));
  __bf16* Fnxtb  = (__bf16*)carve(p, (size_t)SNr * Dh * sizeof(__bf16));
  float*  xn2    = (float*)carve(p, SNr * sizeof(float));
  float*  z2     = (float*)carve(p, Mi * sizeof(float));
  float*  Kmm    = (float*)carve(p, Mi * Mi * sizeof(float));
  float*  LinvF  = (float*)carve(p, Mi * Mi * sizeof(float));
  __bf16* Linvb  = (__bf16*)carve(p, Mi * Mi * sizeof(__bf16));
  __bf16* Zb     = (__bf16*)carve(p, Mi * Dh * sizeof(__bf16));
  __bf16* qmub   = (__bf16*)carve(p, Mi * Dh * sizeof(__bf16));
  __bf16* Wb     = (__bf16*)carve(p, Dh * Dh * sizeof(__bf16));
  __bf16* Kt     = (__bf16*)carve(p, (size_t)Mi * SNr * sizeof(__bf16));
  __bf16* Aout   = (__bf16*)carve(p, (size_t)Mi * SNr * sizeof(__bf16));
  float*  fvar0  = (float*)carve(p, SNr * sizeof(float));
  float*  fvarq  = (float*)carve(p, (size_t)SNr * Dh * sizeof(float));
  __bf16* G      = (__bf16*)carve(p, (size_t)Dh * Mi * Mi * sizeof(__bf16));
  float*  Fmean  = (float*)carve(p, SNr * sizeof(float));

  dim3 b256(256), b128(128), b16(16, 16);
  int wavesNM = (SNr / 16) * (Mi / 16);   // 10240 waves
  int nSnT = SNr / 16;                    // 1280 sn tiles

  // F = tile(X) over samples
  k_tileX<<<(SNr * Dh) / 256, b256, 0, stream>>>(X, Fcur, Fcurb);

  for (int l = 0; l < 2; ++l) {
    const float* Zl = Z_h + (size_t)l * Mi * Dh;
    const float* ql = qmu_h + (size_t)l * Mi * Dh;
    const float* sl = qsq_h + (size_t)l * Mi * Mi * Dh;
    const float* Wl = W_h + (size_t)l * Dh * Dh;

    k_kmm<<<dim3(Mi / 16, Mi / 16), b16, 0, stream>>>(Zl, Kmm, 1e-5f + 1e-6f);
    k_cholesky<<<1, 128, 0, stream>>>(Kmm);
    k_trinv<<<1, 128, 0, stream>>>(Kmm, LinvF);
    k_convert<<<64, b256, 0, stream>>>(LinvF, Linvb, Mi * Mi);
    k_convert<<<32, b256, 0, stream>>>(Zl, Zb, Mi * Dh);
    k_convert<<<32, b256, 0, stream>>>(ql, qmub, Mi * Dh);
    k_convert<<<16, b256, 0, stream>>>(Wl, Wb, Dh * Dh);
    k_rownorm<<<SNr / 256, b256, 0, stream>>>(Fcur, xn2, SNr);
    k_rownorm<<<1, 128, 0, stream>>>(Zl, z2, Mi);

    k_knm<<<wavesNM / 4, b128, 0, stream>>>(Fcurb, Zb, xn2, z2, Kt);
    k_Asolve<<<nSnT, b128, 0, stream>>>(Linvb, Kt, Aout);
    k_fvar0<<<SNr / 256, b256, 0, stream>>>(Aout, fvar0, 1.0f + 1e-5f);
    k_make_G<<<1024, b256, 0, stream>>>(sl, G, Dh);
    k_fvarq<<<dim3(nSnT, Dh / 4), b128, 0, stream>>>(G, Aout, fvarq, Dh);
    k_mean_sample<<<nSnT, b128, 0, stream>>>(
        Aout, qmub, Fcurb, Wb, fvar0, fvarq, Fnxt, Fnxtb,
        0xA341u + (unsigned)l * 0x9E3779B9u);

    float* tf = Fcur; Fcur = Fnxt; Fnxt = tf;
    __bf16* tb = Fcurb; Fcurb = Fnxtb; Fnxtb = tb;
  }

  // final layer (white_var = 0, zero mean fn, Dout = 1)
  k_kmm<<<dim3(Mi / 16, Mi / 16), b16, 0, stream>>>(Z_f, Kmm, 1e-6f);
  k_cholesky<<<1, 128, 0, stream>>>(Kmm);
  k_trinv<<<1, 128, 0, stream>>>(Kmm, LinvF);
  k_convert<<<64, b256, 0, stream>>>(LinvF, Linvb, Mi * Mi);
  k_convert<<<32, b256, 0, stream>>>(Z_f, Zb, Mi * Dh);
  k_rownorm<<<SNr / 256, b256, 0, stream>>>(Fcur, xn2, SNr);
  k_rownorm<<<1, 128, 0, stream>>>(Z_f, z2, Mi);

  k_knm<<<wavesNM / 4, b128, 0, stream>>>(Fcurb, Zb, xn2, z2, Kt);
  k_Asolve<<<nSnT, b128, 0, stream>>>(Linvb, Kt, Aout);
  k_fvar0<<<SNr / 256, b256, 0, stream>>>(Aout, fvar0, 1.0f);
  k_make_G<<<64, b256, 0, stream>>>(qsq_f, G, 1);
  k_fvarq<<<dim3(nSnT, 1), b128, 0, stream>>>(G, Aout, fvarq, 1);   // -> fvarq[sn]
  k_matvec<<<SNr / 256, b256, 0, stream>>>(Aout, qmu_f, Fmean);

  k_ve<<<80, b256, 0, stream>>>(Y, Fmean, fvar0, fvarq, pv);
  k_kl<<<64, b256, 0, stream>>>(qmu_h, qsq_h, Dh, pk);
  k_kl<<<64, b256, 0, stream>>>(qmu_h + (size_t)Mi * Dh, qsq_h + (size_t)Mi * Mi * Dh, Dh, pk + 64);
  k_kl<<<64, b256, 0, stream>>>(qmu_f, qsq_f, 1, pk + 128);
  k_finalize<<<1, 1, 0, stream>>>(pv, pk, out);
}
